// DockingScoreModel_49435073577011
// MI455X (gfx1250) — matrix operand order repaired and checked by reading, not compiled, not run on hardware
//
#include <hip/hip_runtime.h>
#include <hip/hip_bf16.h>

// ---------------------------------------------------------------------------
// MI455X (gfx1250) implementation of the equivariant docking-score GNN.
//  * All edge/node MLPs run on v_wmma_f32_16x16x32_f16 (f16 in, f32 accum).
//  * Tensor-product (Clebsch-Gordan) paths are unrolled VALU code.
//  * segment_mean via f32 global atomics + per-node counts.
// ---------------------------------------------------------------------------

typedef _Float16 h8  __attribute__((ext_vector_type(8)));
typedef _Float16 v16h __attribute__((ext_vector_type(16)));
typedef float    v8f  __attribute__((ext_vector_type(8)));

#define DEV __device__ __forceinline__

// ----------------------------- small helpers -------------------------------

__global__ void k_zero_f32(float* p, long n) {
    long i = (long)blockIdx.x * blockDim.x + threadIdx.x;
    if (i < n) p[i] = 0.0f;
}
__global__ void k_zero_i32(int* p, long n) {
    long i = (long)blockIdx.x * blockDim.x + threadIdx.x;
    if (i < n) p[i] = 0;
}
__global__ void k_count(const int* __restrict__ idx, int n, int* __restrict__ cnt) {
    int i = blockIdx.x * blockDim.x + threadIdx.x;
    if (i < n) atomicAdd(&cnt[idx[i]], 1);
}

// Transpose f32 weight [K x N] -> f16 [N x Kpad] (K-major rows, zero padded).
__global__ void k_wt(const float* __restrict__ W, int K, int N, int Kpad,
                     _Float16* __restrict__ Wt) {
    int i = blockIdx.x * blockDim.x + threadIdx.x;
    if (i >= N * Kpad) return;
    int n = i / Kpad, k = i - n * Kpad;
    Wt[i] = (k < K) ? (_Float16)W[(size_t)k * N + n] : (_Float16)0.0f;
}

// ----------------------------- WMMA GEMM -----------------------------------
// D[E x N] = act(A[E x K(f16)] * W[K x N] + bias), W given transposed (N x K).
// One wave per 16x16 output tile; K stepped by 32.
// mode 0: f32 out, 1: relu -> f16 out, 2: f16 out.
__global__ __launch_bounds__(256) void k_gemm(
    const _Float16* __restrict__ A, int lda,
    const _Float16* __restrict__ Bt, int ldb,
    const float* __restrict__ bias,
    int E, int K, int N, void* __restrict__ out, int ldo, int mode)
{
    const int lane = threadIdx.x & 31;
    const int wid  = blockIdx.x * (blockDim.x >> 5) + (threadIdx.x >> 5);
    const int ntiles = N >> 4;
    const int mt = wid / ntiles;
    const int nt = wid - mt * ntiles;
    if (mt * 16 >= E) return;
    const int hl = lane & 15;
    int arow = mt * 16 + hl; if (arow >= E) arow = E - 1;
    // A fragment: lane<16 -> K {kk..kk+7, kk+16..kk+23}; lane>=16 -> +8.
    const _Float16* ap = A  + (size_t)arow * lda + ((lane < 16) ? 0 : 8);
    // B fragment: lane<16 -> K {kk..kk+7, kk+16..kk+23} of column n; lane>=16 -> +8.
    const _Float16* bp = Bt + (size_t)(nt * 16 + hl) * ldb + ((lane < 16) ? 0 : 8);
    v8f c = {0.f, 0.f, 0.f, 0.f, 0.f, 0.f, 0.f, 0.f};
    for (int kk = 0; kk < K; kk += 32) {
        h8 a0 = *(const h8*)(ap + kk);
        h8 a1 = *(const h8*)(ap + kk + 16);
        h8 b0 = *(const h8*)(bp + kk);
        h8 b1 = *(const h8*)(bp + kk + 16);
        v16h av, bv;
#pragma unroll
        for (int i = 0; i < 8; ++i) { av[i] = a0[i]; av[8 + i] = a1[i];
                                      bv[i] = b0[i]; bv[8 + i] = b1[i]; }
        c = __builtin_amdgcn_wmma_f32_16x16x32_f16(false, av, false, bv,
                                                   (short)0, c, false, false);
    }
    const int col = nt * 16 + hl;
    const float bb = bias ? bias[col] : 0.0f;
#pragma unroll
    for (int r = 0; r < 8; ++r) {
        int row = mt * 16 + ((lane < 16) ? r : (8 + r));
        if (row >= E) continue;
        float v = c[r] + bb;
        if (mode == 1 && v < 0.0f) v = 0.0f;
        if (mode == 0) ((float*)out)[(size_t)row * ldo + col] = v;
        else           ((_Float16*)out)[(size_t)row * ldo + col] = (_Float16)v;
    }
}

// ------------------------- featurization kernels ---------------------------

DEV void dev_sph(float vx, float vy, float vz, float* sh) {
    float r = sqrtf(vx * vx + vy * vy + vz * vz);
    float inv = 1.0f / fmaxf(r, 1e-9f);
    float x = vx * inv, y = vy * inv, z = vz * inv;
    const float s3 = 1.7320508075688772f, s15 = 3.872983346207417f, s5 = 2.23606797749979f;
    sh[0] = 1.0f; sh[1] = s3 * x; sh[2] = s3 * y; sh[3] = s3 * z;
    sh[4] = s15 * x * y; sh[5] = s15 * y * z; sh[6] = 0.5f * s5 * (3.0f * z * z - 1.0f);
    sh[7] = s15 * x * z; sh[8] = 0.5f * s15 * (x * x - y * y);
}
DEV void dev_smear(float d, float stop, _Float16* out) {
    float step = stop / 31.0f;
    float coeff = -0.5f / (step * step);
#pragma unroll
    for (int j = 0; j < 32; ++j) {
        float t = d - step * (float)j;
        out[j] = (_Float16)__expf(coeff * t * t);
    }
}

// Edge features: vec = posB[idxB] - posA[idxA]; X row = [attr | smear32 | 0-pad];
// sh written to persistent array at global edge index.
__global__ void k_edge_feat(const float* __restrict__ posA, const int* __restrict__ idxA,
                            const float* __restrict__ posB, const int* __restrict__ idxB,
                            const float* __restrict__ attr, int nattr, float stop,
                            int e0, int nE, _Float16* __restrict__ X, int ldx, int kz,
                            float* __restrict__ shfull)
{
    int e = blockIdx.x * blockDim.x + threadIdx.x;
    if (e >= nE) return;
    int ge = e0 + e;
    int a = idxA[ge], b = idxB[ge];
    float vx = posB[(size_t)b * 3 + 0] - posA[(size_t)a * 3 + 0];
    float vy = posB[(size_t)b * 3 + 1] - posA[(size_t)a * 3 + 1];
    float vz = posB[(size_t)b * 3 + 2] - posA[(size_t)a * 3 + 2];
    dev_sph(vx, vy, vz, shfull + (size_t)ge * 9);
    _Float16* row = X + (size_t)e * ldx;
    for (int j = 0; j < nattr; ++j) row[j] = (_Float16)attr[(size_t)ge * nattr + j];
    float d = sqrtf(vx * vx + vy * vy + vz * vz);
    dev_smear(d, stop, row + nattr);
    for (int j = nattr + 32; j < kz; ++j) row[j] = (_Float16)0.0f;
}

// X row = [ef(16) | nfA[idxA](16) | nfB[idxB](16) | 0(16)]   (f16, ldx=64)
__global__ void k_concat3(const float* __restrict__ ef,
                          const float* __restrict__ nfA, const int* __restrict__ idxA,
                          const float* __restrict__ nfB, const int* __restrict__ idxB,
                          int e0, int nE, _Float16* __restrict__ X, int ldx)
{
    int i = blockIdx.x * blockDim.x + threadIdx.x;
    if (i >= nE * 16) return;
    int e = i >> 4, c = i & 15, ge = e0 + e;
    _Float16* row = X + (size_t)e * ldx;
    row[c]      = (_Float16)ef[(size_t)ge * 16 + c];
    row[16 + c] = (_Float16)nfA[(size_t)idxA[ge] * 48 + c];
    row[32 + c] = (_Float16)nfB[(size_t)idxB[ge] * 48 + c];
    row[48 + c] = (_Float16)0.0f;
}

struct Tabs9 { const float* t[9]; };
__global__ void k_embed(Tabs9 tabs, int ntab, const int* __restrict__ xc, int ncat,
                        int n, float* __restrict__ out, int ldo)
{
    int i = blockIdx.x * blockDim.x + threadIdx.x;
    if (i >= n * 16) return;
    int node = i >> 4, c = i & 15;
    float s = 0.0f;
    for (int t = 0; t < ntab; ++t) {
        int id = xc[(size_t)node * ncat + t];
        s += tabs.t[t][(size_t)id * 16 + c];
    }
    out[(size_t)node * ldo + c] = s;
}

__global__ void k_center_accum(const float* __restrict__ pos, const int* __restrict__ batch,
                               int n, float* __restrict__ centers, int* __restrict__ cntg)
{
    int i = blockIdx.x * blockDim.x + threadIdx.x;
    if (i >= n) return;
    int b = batch[i];
    atomicAdd(&centers[b * 3 + 0], pos[(size_t)i * 3 + 0]);
    atomicAdd(&centers[b * 3 + 1], pos[(size_t)i * 3 + 1]);
    atomicAdd(&centers[b * 3 + 2], pos[(size_t)i * 3 + 2]);
    atomicAdd(&cntg[b], 1);
}

__global__ void k_center_feat(const float* __restrict__ pos, const int* __restrict__ batch,
                              const float* __restrict__ centers, const int* __restrict__ cntg,
                              int n, _Float16* __restrict__ X, int ldx, float* __restrict__ sh)
{
    int i = blockIdx.x * blockDim.x + threadIdx.x;
    if (i >= n) return;
    int b = batch[i];
    float inv = 1.0f / fmaxf((float)cntg[b], 1.0f);
    float vx = pos[(size_t)i * 3 + 0] - centers[b * 3 + 0] * inv;
    float vy = pos[(size_t)i * 3 + 1] - centers[b * 3 + 1] * inv;
    float vz = pos[(size_t)i * 3 + 2] - centers[b * 3 + 2] * inv;
    dev_sph(vx, vy, vz, sh + (size_t)i * 9);
    float d = sqrtf(vx * vx + vy * vy + vz * vz);
    dev_smear(d, 30.0f, X + (size_t)i * ldx);
}

__global__ void k_concat_final(const float* __restrict__ cmlp, const float* __restrict__ ln,
                               int n, _Float16* __restrict__ X, int ldx)
{
    int i = blockIdx.x * blockDim.x + threadIdx.x;
    if (i >= n * 16) return;
    int node = i >> 4, c = i & 15;
    _Float16* row = X + (size_t)node * ldx;
    row[c]      = (_Float16)cmlp[(size_t)node * 16 + c];
    row[16 + c] = (_Float16)ln[(size_t)node * 48 + c];
}

__global__ void k_combine(float* __restrict__ nf, int n, int old_dim, int new_dim,
                          const float* __restrict__ accA, const int* __restrict__ cntA,
                          const float* __restrict__ accB, const int* __restrict__ cntB)
{
    int i = blockIdx.x * blockDim.x + threadIdx.x;
    if (i >= n * new_dim) return;
    int node = i / new_dim, c = i - node * new_dim;
    float v = (c < old_dim) ? nf[(size_t)node * 48 + c] : 0.0f;
    v += accA[(size_t)node * 48 + c] / fmaxf((float)cntA[node], 1.0f);
    v += accB[(size_t)node * 48 + c] / fmaxf((float)cntB[node], 1.0f);
    nf[(size_t)node * 48 + c] = v;
}

__global__ void k_output(const float* __restrict__ gacc, const int* __restrict__ cntg,
                         int G, float* __restrict__ out)
{
    int i = blockIdx.x * blockDim.x + threadIdx.x;
    if (i >= G * 3) return;
    int g = i / 3, c = i - g * 3;
    float inv = 1.0f / fmaxf((float)cntg[g], 1.0f);
    const float* gp = gacc + (size_t)g * 12;
    out[g * 3 + c]         = (gp[c] + gp[6 + c]) * inv;    // tr_pred
    out[G * 3 + g * 3 + c] = (gp[3 + c] + gp[9 + c]) * inv; // rot_pred
}

// ------------------- equivariant tensor product (unrolled) -----------------

struct TPPath { int o1, mu1, l1, o2, l2, o3, mu3, l3, wo; float alpha; };

constexpr TPPath TP0[2] = {
    {0, 16, 0, 0, 0, 0, 16, 0, 0,   0.25f},
    {0, 16, 0, 1, 1, 16, 4, 1, 256, 0.25f},
};
constexpr TPPath TP1[6] = {
    {0, 16, 0, 0, 0, 0, 16, 0, 0,    0.22360679774997896f},
    {0, 16, 0, 1, 1, 16, 4, 1, 256,  0.20412414523193154f},
    {16, 4, 1, 0, 0, 16, 4, 1, 320,  0.20412414523193154f},
    {16, 4, 1, 1, 1, 0, 16, 0, 336,  0.22360679774997896f},
    {16, 4, 1, 1, 1, 28, 4, 1, 400,  0.5f},
    {16, 4, 1, 4, 2, 16, 4, 1, 416,  0.20412414523193154f},
};
constexpr TPPath TPF[7] = {
    {0, 16, 0, 1, 1, 0, 2, 1, 0,   0.1889822365046136f},
    {16, 4, 1, 0, 0, 0, 2, 1, 32,  0.1889822365046136f},
    {16, 4, 1, 1, 1, 6, 2, 1, 40,  0.25f},
    {16, 4, 1, 4, 2, 0, 2, 1, 48,  0.1889822365046136f},
    {28, 4, 1, 0, 0, 6, 2, 1, 56,  0.25f},
    {28, 4, 1, 1, 1, 0, 2, 1, 64,  0.1889822365046136f},
    {28, 4, 1, 4, 2, 6, 2, 1, 72,  0.25f},
};

DEV void tp_contract(int l1, int l2, int l3, const float* x, const float* s, float* t) {
    const float R3 = 0.5773502691896258f;   // 1/sqrt(3)
    const float R6 = 0.4082482904638631f;   // 1/sqrt(6)
    const float S2 = 0.7071067811865476f;   // 1/sqrt(2)
    const float R5 = 0.4472135954999579f;   // 1/sqrt(5)
    if (l1 == 0 && l2 == 0) { t[0] = x[0] * s[0]; }
    else if (l1 == 0 && l2 == 1) { t[0] = R3 * x[0] * s[0]; t[1] = R3 * x[0] * s[1]; t[2] = R3 * x[0] * s[2]; }
    else if (l1 == 1 && l2 == 0) { t[0] = R3 * x[0] * s[0]; t[1] = R3 * x[1] * s[0]; t[2] = R3 * x[2] * s[0]; }
    else if (l1 == 1 && l2 == 1 && l3 == 0) { t[0] = R3 * (x[0] * s[0] + x[1] * s[1] + x[2] * s[2]); }
    else if (l1 == 1 && l2 == 1 && l3 == 1) {
        t[0] = R6 * (x[1] * s[2] - x[2] * s[1]);
        t[1] = R6 * (x[2] * s[0] - x[0] * s[2]);
        t[2] = R6 * (x[0] * s[1] - x[1] * s[0]);
    } else { // (1,2,1)
        t[0] = R5 * (s[0] * S2 * x[1] - s[2] * R6 * x[0] + s[3] * S2 * x[2] + s[4] * S2 * x[0]);
        t[1] = R5 * (s[0] * S2 * x[0] + s[1] * S2 * x[2] - s[2] * R6 * x[1] - s[4] * S2 * x[1]);
        t[2] = R5 * (s[1] * S2 * x[1] + s[2] * 2.0f * R6 * x[2] + s[3] * S2 * x[0]);
    }
}

template <int LAYER>
__global__ __launch_bounds__(256) void k_tp(
    int nE, const int* __restrict__ gidx, const float* __restrict__ nf, int ldn,
    const float* __restrict__ sh, const _Float16* __restrict__ w, int ldw,
    const int* __restrict__ sidx, float* __restrict__ acc, int ldacc)
{
    int e = blockIdx.x * blockDim.x + threadIdx.x;
    if (e >= nE) return;
    constexpr int NP   = (LAYER == 0) ? 2 : (LAYER == 1) ? 6 : 7;
    constexpr int IDIM = (LAYER == 0) ? 16 : (LAYER == 1) ? 28 : 40;
    constexpr int ODIM = (LAYER == 0) ? 28 : (LAYER == 1) ? 40 : 12;
    constexpr const TPPath* P = (LAYER == 0) ? TP0 : (LAYER == 1) ? TP1 : TPF;

    int gr = gidx ? gidx[e] : e;
    const float* xr = nf + (size_t)gr * ldn;
    float x[IDIM];
#pragma unroll
    for (int i = 0; i < IDIM; ++i) x[i] = xr[i];
    float s9[9];
#pragma unroll
    for (int j = 0; j < 9; ++j) s9[j] = sh[(size_t)e * 9 + j];
    float r[ODIM];
#pragma unroll
    for (int i = 0; i < ODIM; ++i) r[i] = 0.0f;
    const _Float16* wr0 = w + (size_t)e * ldw;
#pragma unroll
    for (int p = 0; p < NP; ++p) {
        const int d1 = 2 * P[p].l1 + 1;
        const int d3 = 2 * P[p].l3 + 1;
#pragma unroll
        for (int u = 0; u < P[p].mu1; ++u) {
            float t[3];
            tp_contract(P[p].l1, P[p].l2, P[p].l3, x + P[p].o1 + u * d1, s9 + P[p].o2, t);
            const _Float16* wrow = wr0 + P[p].wo + u * P[p].mu3;
#pragma unroll
            for (int v = 0; v < P[p].mu3; ++v) {
                float wv = (float)wrow[v] * P[p].alpha;
#pragma unroll
                for (int k = 0; k < d3; ++k) r[P[p].o3 + v * d3 + k] += wv * t[k];
            }
        }
    }
    int dst = sidx[e];
#pragma unroll
    for (int c = 0; c < ODIM; ++c) atomicAdd(&acc[(size_t)dst * ldacc + c], r[c]);
}

// ------------------------------- host side ---------------------------------

static inline int imin(int a, int b) { return a < b ? a : b; }

extern "C" void kernel_launch(void* const* d_in, const int* in_sizes, int n_in,
                              void* d_out, int out_size, void* d_ws, size_t ws_size,
                              hipStream_t stream)
{
    // Params flattened in pytree order (sorted dict keys; 'b' < 'w'); leaves:
    //  0-3 center_edge_mlp | 4-11 convs.l2r[0..1] | 12-19 convs.lig | 20-27 convs.r2l
    //  28-35 convs.rec | 36-39 cross_edge_mlp | 40-43 final_fc | 44-47 lig_edge_mlp
    //  48-56 lig_emb[0..8] | 57-60 rec_edge_mlp | 61 rec_emb[0]
    //  then: 62 lig_x 63 lig_pos 64 lig_edge_index 65 lig_edge_attr 66 rec_x
    //        67 rec_pos 68 rec_edge_index 69 cross_lig 70 cross_rec 71 lig_batch 72 G
    if (n_in < 73) return;

    const int n_lig   = in_sizes[63] / 3;
    const int n_rec   = in_sizes[67] / 3;
    const int e_lig   = in_sizes[64] / 2;
    const int e_rec   = in_sizes[68] / 2;
    const int e_cross = in_sizes[69];
    const int G       = out_size / 6;
    if (G <= 0) return;

    const int*   lig_x   = (const int*)d_in[62];
    const float* lig_pos = (const float*)d_in[63];
    const int*   lsrc    = (const int*)d_in[64];
    const int*   ldstp   = lsrc + e_lig;
    const float* lig_ea  = (const float*)d_in[65];
    const int*   rec_x   = (const int*)d_in[66];
    const float* rec_pos = (const float*)d_in[67];
    const int*   rsrc    = (const int*)d_in[68];
    const int*   rdstp   = rsrc + e_rec;
    const int*   clig    = (const int*)d_in[69];
    const int*   crec    = (const int*)d_in[70];
    const int*   lbatch  = (const int*)d_in[71];

    // ---------------- workspace arena ----------------
    char* base = (char*)d_ws;
    size_t off = 0;
    auto alloc = [&](size_t bytes) -> void* {
        off = (off + 255) & ~(size_t)255;
        void* p = base + off;
        off += bytes;
        return p;
    };

    float* ln   = (float*)alloc((size_t)n_lig * 48 * 4);
    float* rn   = (float*)alloc((size_t)n_rec * 48 * 4);
    float* lea  = (float*)alloc((size_t)e_lig * 16 * 4);
    float* rea  = (float*)alloc((size_t)e_rec * 16 * 4);
    float* cea  = (float*)alloc((size_t)e_cross * 16 * 4);
    float* lsh  = (float*)alloc((size_t)e_lig * 9 * 4);
    float* rsh  = (float*)alloc((size_t)e_rec * 9 * 4);
    float* csh  = (float*)alloc((size_t)e_cross * 9 * 4);
    float* esh  = (float*)alloc((size_t)n_lig * 9 * 4);
    int* cnt_ls = (int*)alloc((size_t)n_lig * 4);
    int* cnt_cl = (int*)alloc((size_t)n_lig * 4);
    int* cnt_rs = (int*)alloc((size_t)n_rec * 4);
    int* cnt_cr = (int*)alloc((size_t)n_rec * 4);
    int* cntg   = (int*)alloc((size_t)G * 4);
    float* accAl = (float*)alloc((size_t)n_lig * 48 * 4);
    float* accBl = (float*)alloc((size_t)n_lig * 48 * 4);
    float* accAr = (float*)alloc((size_t)n_rec * 48 * 4);
    float* accBr = (float*)alloc((size_t)n_rec * 48 * 4);
    float* centers = (float*)alloc((size_t)G * 3 * 4);
    float* gacc    = (float*)alloc((size_t)G * 12 * 4);
    float* cmlp    = (float*)alloc((size_t)n_lig * 16 * 4);
    _Float16* Xn   = (_Float16*)alloc((size_t)n_lig * 32 * 2);
    _Float16* Hn   = (_Float16*)alloc((size_t)n_lig * 32 * 2);
    _Float16* wfin = (_Float16*)alloc((size_t)n_lig * 80 * 2);

    auto LB = [](long n) { return dim3((unsigned)((n + 255) / 256)); };
    auto zf32 = [&](float* p, long n) { k_zero_f32<<<LB(n), 256, 0, stream>>>(p, n); };
    auto zi32 = [&](int* p, long n)   { k_zero_i32<<<LB(n), 256, 0, stream>>>(p, n); };

    // ---------------- weight prep (f32 -> transposed f16) ----------------
    struct MlpW { const float *b1, *w1, *b2, *w2; _Float16 *w1t, *w2t; int k1p, n1, k2p, n2; };
    auto prep = [&](int leaf, int k1, int n1, int k2, int n2) -> MlpW {
        MlpW m;
        m.b1 = (const float*)d_in[leaf + 0]; m.w1 = (const float*)d_in[leaf + 1];
        m.b2 = (const float*)d_in[leaf + 2]; m.w2 = (const float*)d_in[leaf + 3];
        m.n1 = n1; m.n2 = n2;
        m.k1p = (k1 + 31) & ~31; m.k2p = (k2 + 31) & ~31;
        m.w1t = (_Float16*)alloc((size_t)n1 * m.k1p * 2);
        m.w2t = (_Float16*)alloc((size_t)n2 * m.k2p * 2);
        k_wt<<<LB((long)n1 * m.k1p), 256, 0, stream>>>(m.w1, k1, n1, m.k1p, m.w1t);
        k_wt<<<LB((long)n2 * m.k2p), 256, 0, stream>>>(m.w2, k2, n2, m.k2p, m.w2t);
        return m;
    };
    MlpW mCenter = prep(0,  32, 16, 16, 16);
    MlpW mL2r0   = prep(4,  48, 48, 48, 320);
    MlpW mLig0   = prep(12, 48, 48, 48, 320);
    MlpW mLig1   = prep(16, 48, 48, 48, 432);
    MlpW mR2l0   = prep(20, 48, 48, 48, 320);
    MlpW mR2l1   = prep(24, 48, 48, 48, 432);
    MlpW mRec0   = prep(28, 48, 48, 48, 320);
    MlpW mCross  = prep(36, 32, 16, 16, 16);
    MlpW mFinal  = prep(40, 32, 32, 32, 80);
    MlpW mLigE   = prep(44, 36, 16, 16, 16);
    MlpW mRecE   = prep(57, 32, 16, 16, 16);

    // ---------------- chunk buffers (sized to remaining ws) ----------------
    int CHUNK = 32768;
    {
        size_t per_row = (size_t)(64 + 64 + 448) * 2;
        size_t remain = (ws_size > off) ? (ws_size - off) : 0;
        while (CHUNK > 16 && (size_t)CHUNK * per_row + 4096 > remain) CHUNK >>= 1;
        if ((size_t)CHUNK * per_row + 4096 > remain) return;
    }
    _Float16* Xch = (_Float16*)alloc((size_t)CHUNK * 64 * 2);
    _Float16* Hch = (_Float16*)alloc((size_t)CHUNK * 64 * 2);
    _Float16* Wch = (_Float16*)alloc((size_t)CHUNK * 448 * 2);

    auto gemm = [&](const _Float16* A, int lda, const _Float16* Bt, int ldb,
                    const float* bias, int E, int K, int N, void* out, int ldo, int mode) {
        long waves = (long)((E + 15) / 16) * (N / 16);
        int blocks = (int)((waves + 7) / 8);
        k_gemm<<<dim3(blocks), 256, 0, stream>>>(A, lda, Bt, ldb, bias, E, K, N, out, ldo, mode);
    };

    // ---------------- node embeddings ----------------
    zf32(ln, (long)n_lig * 48);
    zf32(rn, (long)n_rec * 48);
    {
        Tabs9 t{}; for (int i = 0; i < 9; ++i) t.t[i] = (const float*)d_in[48 + i];
        k_embed<<<LB((long)n_lig * 16), 256, 0, stream>>>(t, 9, lig_x, 9, n_lig, ln, 48);
        Tabs9 tr{}; tr.t[0] = (const float*)d_in[61];
        k_embed<<<LB((long)n_rec * 16), 256, 0, stream>>>(tr, 1, rec_x, 1, n_rec, rn, 48);
    }

    // ---------------- per-node edge counts (seg_mean denominators) --------
    zi32(cnt_ls, n_lig); zi32(cnt_cl, n_lig); zi32(cnt_rs, n_rec); zi32(cnt_cr, n_rec);
    k_count<<<LB(e_lig), 256, 0, stream>>>(lsrc, e_lig, cnt_ls);
    k_count<<<LB(e_cross), 256, 0, stream>>>(clig, e_cross, cnt_cl);
    k_count<<<LB(e_rec), 256, 0, stream>>>(rsrc, e_rec, cnt_rs);
    k_count<<<LB(e_cross), 256, 0, stream>>>(crec, e_cross, cnt_cr);

    // ---------------- edge MLPs (lea/rea/cea) + spherical harmonics -------
    auto run_edge_mlp = [&](int E, const float* posA, const int* iA, const float* posB,
                            const int* iB, const float* attr, int nattr, float stop,
                            const MlpW& m, float* ef_out, float* shfull) {
        for (int c0 = 0; c0 < E; c0 += CHUNK) {
            int nE = imin(CHUNK, E - c0);
            int kz = (nattr > 0) ? 64 : 32;
            k_edge_feat<<<LB(nE), 256, 0, stream>>>(posA, iA, posB, iB, attr, nattr,
                                                    stop, c0, nE, Xch, 64, kz, shfull);
            gemm(Xch, 64, m.w1t, m.k1p, m.b1, nE, m.k1p, 16, Hch, 32, 1);
            gemm(Hch, 32, m.w2t, m.k2p, m.b2, nE, m.k2p, 16, ef_out + (size_t)c0 * 16, 16, 0);
        }
    };
    run_edge_mlp(e_lig,   lig_pos, lsrc, lig_pos, ldstp, lig_ea, 4, 5.0f,   mLigE, lea, lsh);
    run_edge_mlp(e_rec,   rec_pos, rsrc, rec_pos, rdstp, nullptr, 0, 30.0f, mRecE, rea, rsh);
    run_edge_mlp(e_cross, lig_pos, clig, rec_pos, crec,  nullptr, 0, 250.0f, mCross, cea, csh);

    // ---------------- conv pipeline ----------------
    auto run_conv = [&](int layer, int E, const float* ef,
                        const float* nfA, const int* idxA, const float* nfB, const int* idxB,
                        const int* gidx, const float* nf_gather, const int* sidx,
                        float* acc, const float* shfull, const MlpW& m) {
        for (int c0 = 0; c0 < E; c0 += CHUNK) {
            int nE = imin(CHUNK, E - c0);
            k_concat3<<<LB((long)nE * 16), 256, 0, stream>>>(ef, nfA, idxA, nfB, idxB,
                                                             c0, nE, Xch, 64);
            gemm(Xch, 64, m.w1t, 64, m.b1, nE, 64, 48, Hch, 64, 1);
            gemm(Hch, 64, m.w2t, 64, m.b2, nE, 64, m.n2, Wch, m.n2, 2);
            if (layer == 0)
                k_tp<0><<<LB(nE), 256, 0, stream>>>(nE, gidx + c0, nf_gather, 48,
                        shfull + (size_t)c0 * 9, Wch, m.n2, sidx + c0, acc, 48);
            else
                k_tp<1><<<LB(nE), 256, 0, stream>>>(nE, gidx + c0, nf_gather, 48,
                        shfull + (size_t)c0 * 9, Wch, m.n2, sidx + c0, acc, 48);
        }
    };

    // Layer 0: lig-intra, rec->lig, rec-intra, lig->rec
    zf32(accAl, (long)n_lig * 48); zf32(accBl, (long)n_lig * 48);
    zf32(accAr, (long)n_rec * 48); zf32(accBr, (long)n_rec * 48);
    run_conv(0, e_lig,   lea, ln, lsrc, ln, ldstp, ldstp, ln, lsrc, accAl, lsh, mLig0);
    run_conv(0, e_cross, cea, ln, clig, rn, crec,  crec,  rn, clig, accBl, csh, mR2l0);
    run_conv(0, e_rec,   rea, rn, rsrc, rn, rdstp, rdstp, rn, rsrc, accAr, rsh, mRec0);
    run_conv(0, e_cross, cea, ln, clig, rn, crec,  clig,  ln, crec, accBr, csh, mL2r0);
    k_combine<<<LB((long)n_lig * 28), 256, 0, stream>>>(ln, n_lig, 16, 28, accAl, cnt_ls, accBl, cnt_cl);
    k_combine<<<LB((long)n_rec * 28), 256, 0, stream>>>(rn, n_rec, 16, 28, accAr, cnt_rs, accBr, cnt_cr);

    // Layer 1: lig-intra, rec->lig
    zf32(accAl, (long)n_lig * 48); zf32(accBl, (long)n_lig * 48);
    run_conv(1, e_lig,   lea, ln, lsrc, ln, ldstp, ldstp, ln, lsrc, accAl, lsh, mLig1);
    run_conv(1, e_cross, cea, ln, clig, rn, crec,  crec,  rn, clig, accBl, csh, mR2l1);
    k_combine<<<LB((long)n_lig * 40), 256, 0, stream>>>(ln, n_lig, 28, 40, accAl, cnt_ls, accBl, cnt_cl);

    // ---------------- readout ----------------
    zf32(centers, (long)G * 3); zi32(cntg, G);
    k_center_accum<<<LB(n_lig), 256, 0, stream>>>(lig_pos, lbatch, n_lig, centers, cntg);
    k_center_feat<<<LB(n_lig), 256, 0, stream>>>(lig_pos, lbatch, centers, cntg, n_lig, Xn, 32, esh);
    gemm(Xn, 32, mCenter.w1t, 32, mCenter.b1, n_lig, 32, 16, Hn, 32, 1);
    gemm(Hn, 32, mCenter.w2t, 32, mCenter.b2, n_lig, 32, 16, cmlp, 16, 0);
    k_concat_final<<<LB((long)n_lig * 16), 256, 0, stream>>>(cmlp, ln, n_lig, Xn, 32);
    gemm(Xn, 32, mFinal.w1t, 32, mFinal.b1, n_lig, 32, 32, Hn, 32, 1);
    gemm(Hn, 32, mFinal.w2t, 32, mFinal.b2, n_lig, 32, 80, wfin, 80, 2);
    zf32(gacc, (long)G * 12);
    k_tp<2><<<LB(n_lig), 256, 0, stream>>>(n_lig, nullptr, ln, 48, esh, wfin, 80,
                                           lbatch, gacc, 12);
    k_output<<<LB((long)G * 3), 256, 0, stream>>>(gacc, cntg, G, (float*)d_out);
}
